// DanceDecoder_61761629716612
// MI455X (gfx1250) — matrix operand order repaired and checked by reading, not compiled, not run on hardware
//
#include <hip/hip_runtime.h>
#include <math.h>

// ---------------------------------------------------------------------------
// Model dimensions (compile-time constants from the reference)
// ---------------------------------------------------------------------------
#define B_    16
#define L_    300
#define NF_   151
#define D_    512
#define CF_   4800
#define DI_   1024          // 2*D
#define DS_   16
#define DTR_  32            // D/16
#define BL_   (B_*L_)       // 4800 token rows

// WMMA types (CDNA5 / gfx1250, wave32)
typedef __attribute__((ext_vector_type(16))) __bf16 v16bf;
typedef __attribute__((ext_vector_type(8)))  float  v8f;

// ---------------------------------------------------------------------------
// Device math helpers
// ---------------------------------------------------------------------------
__device__ __forceinline__ unsigned short f2bf_u(float f) {
  // native RNE f32->bf16 convert (v_cvt_*bf16*; pairs fuse to v_cvt_pk_bf16_f32)
  __bf16 b = (__bf16)f;
  return __builtin_bit_cast(unsigned short, b);
}
__device__ __forceinline__ unsigned pack2bf(float lo, float hi) {
  return (unsigned)f2bf_u(lo) | ((unsigned)f2bf_u(hi) << 16);
}
__device__ __forceinline__ float softplus_f(float x) { return x > 20.f ? x : log1pf(__expf(x)); }
__device__ __forceinline__ float mish_f(float x)     { return x * tanhf(softplus_f(x)); }
__device__ __forceinline__ float gelu_f(float x)     { return 0.5f * x * (1.f + erff(x * 0.70710678118654752f)); }
__device__ __forceinline__ float sigmoid_f(float x)  { return 1.f / (1.f + __expf(-x)); }
__device__ __forceinline__ float silu_f(float x)     { return x * sigmoid_f(x); }

enum { EP_NONE = 0, EP_MISH = 1, EP_GELU = 2, EP_SIGMOID = 3, EP_SOFTPLUS = 4 };

// ---------------------------------------------------------------------------
// Generic GEMM:  C[M,N] = act( A[M,K](fp32) * W[K,N](fp32) + bias ),
// computed with v_wmma_f32_16x16x32_bf16.  Block tile 64x128x32, 256 threads
// (8 wave32s; each wave owns a 2x2 grid of 16x16 D tiles -> 4 WMMAs/K-step,
// 32 WMMAs per block K-step).
//
// Staging is branch-free:
//   - interior blocks: float4 global loads + packed 2xbf16 b32 LDS stores
//   - edge blocks: clamped addresses + v_cndmask zero-select (no exec branches)
// ---------------------------------------------------------------------------
#define TM 64
#define TN 128
#define TK 32
#define LDSP (TK + 2)   // row stride in ushorts (68B, 4-byte aligned, even pad)

__global__ __launch_bounds__(256) void gemm_bf16_wmma(
    const float* __restrict__ A, const float* __restrict__ W,
    const float* __restrict__ bias, float* __restrict__ C,
    int M, int N, int K, int lda, int ldw, int ldc, int ep)
{
  __shared__ unsigned short As[TM][LDSP];   // row-major [m][k], bf16 bits
  __shared__ unsigned short Bs[TN][LDSP];   // transposed [n][k], bf16 bits

  const int tid  = threadIdx.x;
  const int lane = tid & 31, wid = tid >> 5;
  const int m0 = blockIdx.y * TM, n0 = blockIdx.x * TN;
  const int niB = (wid & 3) * 2, miB = (wid >> 2) * 2;   // wave -> 2x2 tile grid

  // Staging coordinates (fixed per thread)
  const int ar = tid >> 2,  acb = (tid & 3) * 8;    // A: row, col-base (8 floats)
  const int bk = tid >> 3,  bnb = (tid & 7) * 16;   // B: k-row, n-base (16 floats)
  const bool interior = (m0 + TM <= M) && (n0 + TN <= N);

  v8f acc00 = {}, acc01 = {}, acc10 = {}, acc11 = {};

  const int ksteps = (K + TK - 1) / TK;
  for (int kt = 0; kt < ksteps; ++kt) {
    const int k0 = kt * TK;

    if (interior && (k0 + TK <= K)) {
      // ---- fast path: vectorized loads, packed stores, zero branches ----
      const float4* Ap = (const float4*)(A + (size_t)(m0 + ar) * lda + k0 + acb);
      float4 a0 = Ap[0], a1 = Ap[1];
      unsigned* Adst = (unsigned*)&As[ar][acb];
      Adst[0] = pack2bf(a0.x, a0.y);
      Adst[1] = pack2bf(a0.z, a0.w);
      Adst[2] = pack2bf(a1.x, a1.y);
      Adst[3] = pack2bf(a1.z, a1.w);

      const float4* Wp = (const float4*)(W + (size_t)(k0 + bk) * ldw + n0 + bnb);
      #pragma unroll
      for (int q = 0; q < 4; ++q) {
        float4 bq = Wp[q];
        Bs[bnb + q * 4 + 0][bk] = f2bf_u(bq.x);
        Bs[bnb + q * 4 + 1][bk] = f2bf_u(bq.y);
        Bs[bnb + q * 4 + 2][bk] = f2bf_u(bq.z);
        Bs[bnb + q * 4 + 3][bk] = f2bf_u(bq.w);
      }
    } else {
      // ---- edge path: clamped addresses + value select (no exec branches) ----
      const int gm = m0 + ar;
      const int gmc = gm < M ? gm : M - 1;
      #pragma unroll
      for (int j = 0; j < 8; ++j) {
        const int gk  = k0 + acb + j;
        const int gkc = gk < K ? gk : K - 1;
        float v = A[(size_t)gmc * lda + gkc];
        v = (gm < M && gk < K) ? v : 0.f;
        As[ar][acb + j] = f2bf_u(v);
      }
      const int gk  = k0 + bk;
      const int gkc = gk < K ? gk : K - 1;
      #pragma unroll
      for (int j = 0; j < 16; ++j) {
        const int gn  = n0 + bnb + j;
        const int gnc = gn < N ? gn : N - 1;
        float v = W[(size_t)gkc * ldw + gnc];
        v = (gk < K && gn < N) ? v : 0.f;
        Bs[bnb + j][bk] = f2bf_u(v);
      }
    }
    __syncthreads();

    // Prefetch next K tile (global_prefetch_b8); clamped addresses.
    if (kt + 1 < ksteps) {
      const int pm = m0 + ar;      const int pmc = pm < M ? pm : M - 1;
      const int pk = k0 + TK;      const int pkc = pk < K ? pk : K - 1;
      __builtin_prefetch(&A[(size_t)pmc * lda + pkc], 0, 1);
      const int qk = k0 + TK + bk; const int qkc = qk < K ? qk : K - 1;
      __builtin_prefetch(&W[(size_t)qkc * ldw + n0], 0, 1);
    }

    // Fragments per the ISA 16-bit layouts (wave32).
    {
      const int mrow = lane & 15, hi = lane >> 4;
      v16bf a0, a1, b0, b1;
      #pragma unroll
      for (int i = 0; i < 16; ++i) {
        const int v = i >> 1;
        // A 16x32: VGPR v holds a K pair; lanes16-31 shift K by 8 (+16 for v>=4)
        const int ka = ((v >> 2) << 4) + (hi << 3) + ((v & 3) << 1) + (i & 1);
        a0[i] = __builtin_bit_cast(__bf16, As[(miB + 0) * 16 + mrow][ka]);
        a1[i] = __builtin_bit_cast(__bf16, As[(miB + 1) * 16 + mrow][ka]);
        // B 32x16: same K order as A (ka); N = lane&15
        b0[i] = __builtin_bit_cast(__bf16, Bs[(niB + 0) * 16 + mrow][ka]);
        b1[i] = __builtin_bit_cast(__bf16, Bs[(niB + 1) * 16 + mrow][ka]);
      }
      acc00 = __builtin_amdgcn_wmma_f32_16x16x32_bf16(false, a0, false, b0,
                                                      (short)0, acc00, false, false);
      acc01 = __builtin_amdgcn_wmma_f32_16x16x32_bf16(false, a0, false, b1,
                                                      (short)0, acc01, false, false);
      acc10 = __builtin_amdgcn_wmma_f32_16x16x32_bf16(false, a1, false, b0,
                                                      (short)0, acc10, false, false);
      acc11 = __builtin_amdgcn_wmma_f32_16x16x32_bf16(false, a1, false, b1,
                                                      (short)0, acc11, false, false);
    }
    __syncthreads();
  }

  // Epilogue per C/D layout: lane&15 = N, VGPR r -> M = (lane>>4)*8 + r.
  const int lane15 = lane & 15, hi = lane >> 4;
  #pragma unroll
  for (int nj = 0; nj < 2; ++nj) {
    const int gn = n0 + (niB + nj) * 16 + lane15;
    if (gn >= N) continue;
    const float bv = bias ? bias[gn] : 0.f;
    #pragma unroll
    for (int mi = 0; mi < 2; ++mi) {
      v8f acc;
      if      (mi == 0 && nj == 0) acc = acc00;
      else if (mi == 0 && nj == 1) acc = acc01;
      else if (mi == 1 && nj == 0) acc = acc10;
      else                         acc = acc11;
      const int mbase = m0 + (miB + mi) * 16 + hi * 8;
      #pragma unroll
      for (int r = 0; r < 8; ++r) {
        const int gm = mbase + r;
        if (gm < M) {
          float v = acc[r] + bv;
          if      (ep == EP_MISH)     v = mish_f(v);
          else if (ep == EP_GELU)     v = gelu_f(v);
          else if (ep == EP_SIGMOID)  v = sigmoid_f(v);
          else if (ep == EP_SOFTPLUS) v = softplus_f(v);
          C[(size_t)gm * ldc + gn] = v;
        }
      }
    }
  }
}

// ---------------------------------------------------------------------------
// LayerNorm over D=512, one wave32 per row (8 rows per 256-thread block).
// ---------------------------------------------------------------------------
__global__ __launch_bounds__(256) void layernorm_kernel(
    const float* __restrict__ x, const float* __restrict__ s,
    const float* __restrict__ b, float* __restrict__ y, int rows)
{
  const int wid = threadIdx.x >> 5, lane = threadIdx.x & 31;
  const int row = blockIdx.x * 8 + wid;
  if (row >= rows) return;
  const float* xr = x + (size_t)row * D_;
  float sum = 0.f, sq = 0.f;
  #pragma unroll
  for (int j = lane; j < D_; j += 32) { float v = xr[j]; sum += v; sq += v * v; }
  #pragma unroll
  for (int m = 16; m; m >>= 1) {
    sum += __shfl_xor(sum, m, 32);
    sq  += __shfl_xor(sq,  m, 32);
  }
  const float mean = sum * (1.f / D_);
  const float var  = sq * (1.f / D_) - mean * mean;
  const float rst  = rsqrtf(var + 1e-5f);
  for (int j = lane; j < D_; j += 32)
    y[(size_t)row * D_ + j] = (xr[j] - mean) * rst * s[j] + b[j];
}

// ---------------------------------------------------------------------------
// Depthwise causal conv (K=4) over time + bias + SiLU.  Reads xm half of the
// packed xz buffer (row stride 2*DI).
// ---------------------------------------------------------------------------
__global__ void dwconv_silu_kernel(const float* __restrict__ xz,
                                   const float* __restrict__ cw,
                                   const float* __restrict__ cb,
                                   float* __restrict__ out)
{
  size_t idx = (size_t)blockIdx.x * blockDim.x + threadIdx.x;
  if (idx >= (size_t)BL_ * DI_) return;
  const int c = (int)(idx % DI_);
  const int row = (int)(idx / DI_);
  const int l = row % L_, b = row / L_;
  float acc = 0.f;
  #pragma unroll
  for (int j = 0; j < 4; ++j) {
    const int ls = l - 3 + j;
    if (ls >= 0) acc += xz[((size_t)(b * L_ + ls)) * (2 * DI_) + c] * cw[c * 4 + j];
  }
  acc += cb[c];
  out[idx] = silu_f(acc);
}

__global__ void aneg_kernel(const float* __restrict__ Alog, float* __restrict__ An) {
  const int i = blockIdx.x * blockDim.x + threadIdx.x;
  if (i < DI_ * DS_) An[i] = -__expf(Alog[i]);
}

// ---------------------------------------------------------------------------
// Selective scan: one thread per (batch, channel); 16 states live in VGPRs,
// sequential over L=300.  16384 threads total; latency-bound, not matrix work.
// ---------------------------------------------------------------------------
__global__ __launch_bounds__(256) void mamba_scan_kernel(
    const float* __restrict__ delta,   // [BL, DI]
    const float* __restrict__ dtbc,    // [BL, 64]  (cols 32..47 = B, 48..63 = C)
    const float* __restrict__ xm,      // [BL, DI]  post conv+silu
    const float* __restrict__ z,       // [BL, *]   stride 2*DI (xz second half)
    const float* __restrict__ An,      // [DI, DS]  -exp(A_log)
    const float* __restrict__ Dp,      // [DI]
    float* __restrict__ y)             // [BL, DI]
{
  const int idx = blockIdx.x * blockDim.x + threadIdx.x;
  if (idx >= B_ * DI_) return;
  const int d = idx % DI_, b = idx / DI_;
  float a[DS_], h[DS_];
  #pragma unroll
  for (int s = 0; s < DS_; ++s) { a[s] = An[d * DS_ + s]; h[s] = 0.f; }
  const float dpv = Dp[d];
  for (int l = 0; l < L_; ++l) {
    const size_t row = (size_t)(b * L_ + l);
    const float dl = delta[row * DI_ + d];
    const float xv = xm[row * DI_ + d];
    const float zv = z[row * (2 * DI_) + d];
    const float* bc = dtbc + row * 64 + 32;
    const float* cc = dtbc + row * 64 + 48;
    const float dx = dl * xv;
    float acc = 0.f;
    #pragma unroll
    for (int s = 0; s < DS_; ++s) {
      h[s] = h[s] * __expf(dl * a[s]) + dx * bc[s];
      acc += h[s] * cc[s];
    }
    y[row * DI_ + d] = (acc + xv * dpv) * silu_f(zv);
  }
}

// ---------------------------------------------------------------------------
// Small elementwise kernels
// ---------------------------------------------------------------------------
__global__ void sinusoidal_kernel(const float* __restrict__ t, float* __restrict__ out) {
  const int idx = blockIdx.x * blockDim.x + threadIdx.x;
  if (idx >= B_ * D_) return;
  const int b = idx / D_, j = idx % D_;
  const float x = t[b] * 1000.f;
  const int hh = j & 255;                       // half = 256
  const float freq = __expf(-9.2103403719761836f * (float)hh / 255.f);
  const float e = x * freq;
  out[idx] = (j < 256) ? __sinf(e) : __cosf(e);
}

__global__ void mishvec_kernel(const float* __restrict__ in, float* __restrict__ out, int n) {
  const int i = blockIdx.x * blockDim.x + threadIdx.x;
  if (i < n) out[i] = mish_f(in[i]);
}
__global__ void addvec_kernel(float* __restrict__ a, const float* __restrict__ b, int n) {
  const int i = blockIdx.x * blockDim.x + threadIdx.x;
  if (i < n) a[i] += b[i];
}
__global__ void gather_emb_kernel(const float* __restrict__ emb, const int* __restrict__ genre,
                                  float* __restrict__ out) {
  const int idx = blockIdx.x * blockDim.x + threadIdx.x;
  if (idx >= B_ * D_) return;
  const int b = idx / D_, c = idx % D_;
  out[idx] = emb[(size_t)genre[b] * D_ + c];
}
__global__ void zc_build_kernel(const float* __restrict__ cond, const float* __restrict__ g,
                                float* __restrict__ zc) {
  size_t idx = (size_t)blockIdx.x * blockDim.x + threadIdx.x;
  if (idx >= (size_t)BL_ * (2 * D_)) return;
  const int c = (int)(idx % (2 * D_));
  const size_t row = idx / (2 * D_);
  const int b = (int)(row / L_);
  zc[idx] = (c < D_) ? cond[row * D_ + c] : g[(size_t)b * D_ + (c - D_)];
}
__global__ void gate_add_kernel(float* __restrict__ x, const float* __restrict__ al,
                                const float* __restrict__ dl, size_t n) {
  size_t i = (size_t)blockIdx.x * blockDim.x + threadIdx.x;
  if (i < n) x[i] += al[i] * dl[i];
}
__global__ void film_add_kernel(float* __restrict__ x, const float* __restrict__ src,
                                const float* __restrict__ e) {
  size_t idx = (size_t)blockIdx.x * blockDim.x + threadIdx.x;
  if (idx >= (size_t)BL_ * D_) return;
  const int c = (int)(idx % D_);
  const size_t row = idx / D_;
  const int b = (int)(row / L_);
  const float sc = e[(size_t)b * (2 * D_) + c];
  const float sh = e[(size_t)b * (2 * D_) + D_ + c];
  x[idx] += (sc + 1.f) * src[idx] + sh;
}
__global__ void flip_copy_kernel(const float* __restrict__ in, float* __restrict__ out) {
  size_t idx = (size_t)blockIdx.x * blockDim.x + threadIdx.x;
  if (idx >= (size_t)BL_ * D_) return;
  const int c = (int)(idx % D_);
  const size_t row = idx / D_;
  const int b = (int)(row / L_), l = (int)(row % L_);
  out[idx] = in[((size_t)(b * L_ + (L_ - 1 - l))) * D_ + c];
}
__global__ void flip_add_kernel(const float* __restrict__ in, float* __restrict__ out) {
  size_t idx = (size_t)blockIdx.x * blockDim.x + threadIdx.x;
  if (idx >= (size_t)BL_ * D_) return;
  const int c = (int)(idx % D_);
  const size_t row = idx / D_;
  const int b = (int)(row / L_), l = (int)(row % L_);
  out[idx] += in[((size_t)(b * L_ + (L_ - 1 - l))) * D_ + c];
}

// ---------------------------------------------------------------------------
// Host-side orchestration
// ---------------------------------------------------------------------------
struct MambaP { const float *A_log, *Dp, *conv_b, *conv_w, *dt_b, *dt_w, *in_w, *out_w, *x_w; };
struct IntraP { const float *ln_b, *ln_s; MambaP mb, mf; };
struct FilmP  { const float *b, *w; };
struct GateP  { const float *a1_b, *a1_w, *a2_b, *a2_w, *d1_b, *d1_w, *d2_b, *d2_w, *emb, *pg_b, *pg_w; };
struct DanceP { FilmP film1, film2; IntraP intra; };
struct MusicP { GateP gate; IntraP intra; };

static inline void gemm(hipStream_t st, const float* A, const float* W, const float* bias,
                        float* C, int M, int N, int K, int lda, int ldw, int ldc, int ep) {
  dim3 g((N + TN - 1) / TN, (M + TM - 1) / TM);
  gemm_bf16_wmma<<<g, 256, 0, st>>>(A, W, bias, C, M, N, K, lda, ldw, ldc, ep);
}
static inline unsigned ew(size_t n) { return (unsigned)((n + 255) / 256); }

struct MambaWs { float *xz, *xm2, *dtbc, *delta, *ym, *aneg; };

static void run_mamba(hipStream_t st, const MambaP& p, const float* in, float* out, MambaWs& w) {
  // in-proj: [BL,512] x [512,2048] -> xz (xm | z)
  gemm(st, in, p.in_w, nullptr, w.xz, BL_, 2 * DI_, D_, D_, 2 * DI_, 2 * DI_, EP_NONE);
  dwconv_silu_kernel<<<ew((size_t)BL_ * DI_), 256, 0, st>>>(w.xz, p.conv_w, p.conv_b, w.xm2);
  // x-proj: [BL,1024] x [1024,64] -> dt|B|C
  gemm(st, w.xm2, p.x_w, nullptr, w.dtbc, BL_, DTR_ + 2 * DS_, DI_, DI_, 64, 64, EP_NONE);
  // delta = softplus(dt @ dt_w + dt_b): [BL,32] x [32,1024]
  gemm(st, w.dtbc, p.dt_w, p.dt_b, w.delta, BL_, DI_, DTR_, 64, DI_, DI_, EP_SOFTPLUS);
  aneg_kernel<<<ew(DI_ * DS_), 256, 0, st>>>(p.A_log, w.aneg);
  mamba_scan_kernel<<<ew(B_ * DI_), 256, 0, st>>>(w.delta, w.dtbc, w.xm2, w.xz + DI_,
                                                  w.aneg, p.Dp, w.ym);
  // out-proj: [BL,1024] x [1024,512]
  gemm(st, w.ym, p.out_w, nullptr, out, BL_, D_, DI_, DI_, D_, D_, EP_NONE);
}

static void run_intra(hipStream_t st, const IntraP& p, const float* in, float* out,
                      MambaWs& w, float* xn, float* xnf, float* tmp) {
  layernorm_kernel<<<(BL_ + 7) / 8, 256, 0, st>>>(in, p.ln_s, p.ln_b, xn, BL_);
  run_mamba(st, p.mf, xn, out, w);
  flip_copy_kernel<<<ew((size_t)BL_ * D_), 256, 0, st>>>(xn, xnf);
  run_mamba(st, p.mb, xnf, tmp, w);
  flip_add_kernel<<<ew((size_t)BL_ * D_), 256, 0, st>>>(tmp, out);
}

extern "C" void kernel_launch(void* const* d_in, const int* in_sizes, int n_in,
                              void* d_out, int out_size, void* d_ws, size_t ws_size,
                              hipStream_t stream) {
  (void)in_sizes; (void)out_size; (void)ws_size;
  const float* x_in       = (const float*)d_in[0];
  const float* cond_embed = (const float*)d_in[1];
  const int*   genre      = (const int*)  d_in[2];
  const float* times      = (const float*)d_in[3];
  const float* interval   = (const float*)d_in[4];

  // --- Param leaves, flattened in sorted-dict-key order (jax tree order) ---
  int p = 5;
  auto nx = [&]() -> const float* {
    const float* r = (p < n_in) ? (const float*)d_in[p] : nullptr; ++p; return r;
  };
  auto rd_mamba = [&](MambaP& m) {
    m.A_log = nx(); m.Dp = nx(); m.conv_b = nx(); m.conv_w = nx();
    m.dt_b = nx(); m.dt_w = nx(); m.in_w = nx(); m.out_w = nx(); m.x_w = nx();
  };
  auto rd_intra = [&](IntraP& i) { i.ln_b = nx(); i.ln_s = nx(); rd_mamba(i.mb); rd_mamba(i.mf); };
  auto rd_film  = [&](FilmP& f)  { f.b = nx(); f.w = nx(); };
  auto rd_gate  = [&](GateP& g)  {
    g.a1_b = nx(); g.a1_w = nx(); g.a2_b = nx(); g.a2_w = nx();
    g.d1_b = nx(); g.d1_w = nx(); g.d2_b = nx(); g.d2_w = nx();
    g.emb = nx(); g.pg_b = nx(); g.pg_w = nx();
  };

  const float* cond_b = nx(); const float* cond_w = nx();
  DanceP dance[8];
  for (int i = 0; i < 8; ++i) { rd_film(dance[i].film1); rd_film(dance[i].film2); rd_intra(dance[i].intra); }
  const float* fin_b = nx(); const float* fin_w = nx();
  const float* i1_b = nx(); const float* i1_w = nx();
  const float* i2_b = nx(); const float* i2_w = nx();
  const float* in_b = nx(); const float* in_w = nx();
  MusicP music[4];
  for (int i = 0; i < 4; ++i) { rd_gate(music[i].gate); rd_intra(music[i].intra); }
  const float* t1_b = nx(); const float* t1_w = nx();
  const float* t2_b = nx(); const float* t2_w = nx();

  // --- Workspace bump allocator over d_ws ---
  char* wp = (char*)d_ws;
  auto alloc = [&](size_t elems) -> float* {
    float* r = (float*)wp; wp += ((elems * sizeof(float)) + 255) & ~(size_t)255; return r;
  };
  float* xbuf  = alloc((size_t)BL_ * D_);
  float* condA = alloc((size_t)BL_ * D_);
  float* condB = alloc((size_t)BL_ * D_);
  float* xn    = alloc((size_t)BL_ * D_);
  float* xnf   = alloc((size_t)BL_ * D_);
  float* mtmp  = alloc((size_t)BL_ * D_);
  float* x1buf = alloc((size_t)BL_ * D_);
  MambaWs mw;
  mw.xz    = alloc((size_t)BL_ * 2 * DI_);      // also aliased as zc (disjoint lifetime)
  mw.xm2   = alloc((size_t)BL_ * DI_);          // also aliased: gate gelu hidden
  mw.dtbc  = alloc((size_t)BL_ * 64);
  mw.delta = alloc((size_t)BL_ * DI_);          // also aliased: gate dl
  mw.ym    = alloc((size_t)BL_ * DI_);          // also aliased: gate al
  mw.aneg  = alloc((size_t)DI_ * DS_);
  float* zc    = mw.xz;                          // [BL, 2D]  (gate phase only)
  float* h1    = mw.xm2;                         // [BL, D]   gelu hidden
  float* dlbuf = mw.delta;                       // [BL, D]
  float* albuf = mw.ym;                          // [BL, D]
  float* sbuf  = alloc((size_t)B_ * D_);
  float* tb1   = alloc((size_t)B_ * 4 * D_);
  float* tbuf  = alloc((size_t)B_ * D_);
  float* ivbuf = alloc((size_t)B_ * D_);
  float* mtbuf = alloc((size_t)B_ * D_);
  float* ebuf  = alloc((size_t)B_ * 2 * D_);
  float* gemb  = alloc((size_t)B_ * D_);
  float* gbuf  = alloc((size_t)B_ * D_);

  // --- 1. input projection: x @ in_w + in_b ---
  gemm(stream, x_in, in_w, in_b, xbuf, BL_, D_, NF_, NF_, D_, D_, EP_NONE);

  // --- 2. time + interval embeddings (tiny M=16 GEMMs) ---
  sinusoidal_kernel<<<ew(B_ * D_), 256, 0, stream>>>(times, sbuf);
  gemm(stream, sbuf, t1_w, t1_b, tb1, B_, 4 * D_, D_, D_, 4 * D_, 4 * D_, EP_MISH);
  gemm(stream, tb1, t2_w, t2_b, tbuf, B_, D_, 4 * D_, 4 * D_, D_, D_, EP_NONE);
  sinusoidal_kernel<<<ew(B_ * D_), 256, 0, stream>>>(interval, sbuf);
  gemm(stream, sbuf, i1_w, i1_b, tb1, B_, 4 * D_, D_, D_, 4 * D_, 4 * D_, EP_MISH);
  gemm(stream, tb1, i2_w, i2_b, ivbuf, B_, D_, 4 * D_, 4 * D_, D_, D_, EP_NONE);
  addvec_kernel<<<ew(B_ * D_), 256, 0, stream>>>(tbuf, ivbuf, B_ * D_);
  mishvec_kernel<<<ew(B_ * D_), 256, 0, stream>>>(tbuf, mtbuf, B_ * D_);

  // --- 3. cond projection: [4800,4800] x [4800,512] (biggest single GEMM) ---
  gemm(stream, cond_embed, cond_w, cond_b, condA, BL_, D_, CF_, CF_, D_, D_, EP_NONE);

  float* cond = condA;
  float* condNext = condB;

  // --- 4. music layers: genre gate + bidirectional intra-mamba ---
  for (int li = 0; li < 4; ++li) {
    const GateP& g = music[li].gate;
    gather_emb_kernel<<<ew(B_ * D_), 256, 0, stream>>>(g.emb, genre, gemb);
    gemm(stream, gemb, g.pg_w, g.pg_b, gbuf, B_, D_, D_, D_, D_, D_, EP_NONE);
    zc_build_kernel<<<ew((size_t)BL_ * 2 * D_), 256, 0, stream>>>(cond, gbuf, zc);
    gemm(stream, zc, g.d1_w, g.d1_b, h1, BL_, D_, 2 * D_, 2 * D_, D_, D_, EP_GELU);
    gemm(stream, h1, g.d2_w, g.d2_b, dlbuf, BL_, D_, D_, D_, D_, D_, EP_NONE);
    gemm(stream, zc, g.a1_w, g.a1_b, h1, BL_, D_, 2 * D_, 2 * D_, D_, D_, EP_GELU);
    gemm(stream, h1, g.a2_w, g.a2_b, albuf, BL_, D_, D_, D_, D_, D_, EP_SIGMOID);
    gate_add_kernel<<<ew((size_t)BL_ * D_), 256, 0, stream>>>(cond, albuf, dlbuf,
                                                              (size_t)BL_ * D_);
    run_intra(stream, music[li].intra, cond, condNext, mw, xn, xnf, mtmp);
    float* t = cond; cond = condNext; condNext = t;
  }

  // --- 5. dance layers: intra-mamba + FiLM(t) + FiLM(cond) ---
  for (int li = 0; li < 8; ++li) {
    run_intra(stream, dance[li].intra, xbuf, x1buf, mw, xn, xnf, mtmp);
    gemm(stream, mtbuf, dance[li].film1.w, dance[li].film1.b, ebuf,
         B_, 2 * D_, D_, D_, 2 * D_, 2 * D_, EP_NONE);
    film_add_kernel<<<ew((size_t)BL_ * D_), 256, 0, stream>>>(xbuf, x1buf, ebuf);
    gemm(stream, mtbuf, dance[li].film2.w, dance[li].film2.b, ebuf,
         B_, 2 * D_, D_, D_, 2 * D_, 2 * D_, EP_NONE);
    film_add_kernel<<<ew((size_t)BL_ * D_), 256, 0, stream>>>(xbuf, cond, ebuf);
  }

  // --- 6. final projection -> d_out [16,300,151] ---
  gemm(stream, xbuf, fin_w, fin_b, (float*)d_out, BL_, NF_, D_, D_, NF_, NF_, EP_NONE);
}
